// AttentionHead_16733192585594
// MI455X (gfx1250) — matrix operand order repaired and checked by reading, not compiled, not run on hardware
//
#include <hip/hip_runtime.h>

// Problem constants (match reference)
#define BATCH 512
#define SEQ   256
#define NEMB  256
#define HS    64
#define DROP_P 0.25f
#define KEEP_INV (1.0f / (1.0f - DROP_P))   // 4/3
// softmax scale (H^-0.5) folded with log2(e) for exp2-based softmax
#define SCALE_LOG2E (0.125f * 1.44269504088896340736f)

typedef __attribute__((ext_vector_type(16))) __bf16 v16bf;
typedef __attribute__((ext_vector_type(2)))  __bf16 v2bf;
typedef __attribute__((ext_vector_type(8)))  float  v8f;

// 8-dword fragment (16 bf16) for WMMA A/B operands
union Frag16 {
    unsigned u[8];
    uint4    q[2];
    v16bf    v;
};

// f32 pair -> packed bf16 dword (lo = a, hi = b).
// Round-to-nearest (ties up) then a single v_perm_b32 byte-select pack.
__device__ __forceinline__ unsigned pkbf(float a, float b) {
#if __has_builtin(__builtin_amdgcn_cvt_pk_bf16_f32)
    union { v2bf v; unsigned u; } cv;
    cv.v = __builtin_amdgcn_cvt_pk_bf16_f32(a, b);
    return cv.u;
#else
    unsigned ua = __float_as_uint(a) + 0x8000u;
    unsigned ub = __float_as_uint(b) + 0x8000u;
    // result bytes: [0]=ua.b2 [1]=ua.b3 [2]=ub.b2 [3]=ub.b3
    return __builtin_amdgcn_perm(ub, ua, 0x07060302u);
#endif
}
__device__ __forceinline__ unsigned short f2bf(float a) {
    // add + (srl folded into d16_hi store by the backend where possible)
    return (unsigned short)((__float_as_uint(a) + 0x8000u) >> 16);
}

__device__ __forceinline__ float fast_exp2(float a) {
#if __has_builtin(__builtin_amdgcn_exp2f)
    return __builtin_amdgcn_exp2f(a);
#else
    return exp2f(a);
#endif
}

__device__ __forceinline__ v8f wmma_bf16(const Frag16& a, const Frag16& b, v8f c) {
    return __builtin_amdgcn_wmma_f32_16x16x32_bf16(false, a.v, false, b.v,
                                                   (short)0, c, false, false);
}

// ---------------------------------------------------------------------------
// Kernel 0: one-time prep — W[256][64] f32 -> W^T[64][256] bf16 for q/k/v.
// 3*16384 elements; grid 192 x 256.
// ---------------------------------------------------------------------------
__global__ __launch_bounds__(256) void prep_kernel(
    const float* __restrict__ Wq, const float* __restrict__ Wk,
    const float* __restrict__ Wv, unsigned short* __restrict__ wt)
{
    const int i = blockIdx.x * 256 + threadIdx.x;      // 0 .. 49151
    const int w = i / (NEMB * HS);
    const int r = i % (NEMB * HS);
    const int c = r >> 6;            // 0..255 (embed dim)
    const int h = r & 63;            // 0..63  (head dim)
    const float* W = (w == 0) ? Wq : (w == 1) ? Wk : Wv;
    wt[w * (NEMB * HS) + h * NEMB + c] = f2bf(W[r]);
}

// ---------------------------------------------------------------------------
// Kernel 1: q = x@Wq, k = x@Wk, v = x@Wv   (outputs bf16; v stored transposed)
// grid = (B*T)/128 workgroups, 256 threads (8 waves), wave = one 16-row tile.
// ---------------------------------------------------------------------------
__global__ __launch_bounds__(256) void qkv_kernel(
    const float* __restrict__ x,
    const unsigned short* __restrict__ wt,  // [3][64][256] bf16 W^T (prepped)
    unsigned short* __restrict__ qo,        // [B*T][64] row-major bf16
    unsigned short* __restrict__ ko,        // [B*T][64] row-major bf16
    unsigned short* __restrict__ vto)       // [B][64][T] bf16 (V transposed)
{
    __shared__ unsigned short sWT[HS * NEMB];   // W^T bf16 [64][256] = 32 KB

    const int tid  = threadIdx.x;
    const int lane = tid & 31;
    const int wave = tid >> 5;
    const int hs   = lane >> 4;      // half-wave select
    const int lm   = lane & 15;
    const int R    = blockIdx.x * 128 + wave * 16;   // global row base

    // Load this tile's x rows once as 8 bf16 A-fragments (K chunks of 32).
    // A layout: lane row = lm; v0..7 = K[16hs..16hs+15].
    Frag16 xa[8];
    const float* xrow = x + (size_t)(R + lm) * NEMB;
    #pragma unroll
    for (int c = 0; c < 8; ++c) {
        #pragma unroll
        for (int g = 0; g < 2; ++g) {
            const float* p = xrow + 32 * c + 16 * hs + 8 * g;
            float4 f0 = *(const float4*)p;
            float4 f1 = *(const float4*)(p + 4);
            xa[c].u[4 * g + 0] = pkbf(f0.x, f0.y);
            xa[c].u[4 * g + 1] = pkbf(f0.z, f0.w);
            xa[c].u[4 * g + 2] = pkbf(f1.x, f1.y);
            xa[c].u[4 * g + 3] = pkbf(f1.z, f1.w);
        }
    }

    const int bb = R / SEQ;   // batch index
    const int t0 = R % SEQ;   // token base within batch

    #pragma unroll
    for (int w = 0; w < 3; ++w) {
        __syncthreads();
        // Stage pre-converted W^T into LDS (straight b128 copy).
        {
            const uint4* gw = (const uint4*)(wt + (size_t)w * NEMB * HS);
            uint4* lw = (uint4*)sWT;
            #pragma unroll
            for (int i = 0; i < (NEMB * HS / 8) / 256; ++i)
                lw[tid + 256 * i] = gw[tid + 256 * i];
        }
        __syncthreads();

        #pragma unroll
        for (int nt = 0; nt < 4; ++nt) {          // 4 n-tiles of 16 over H=64
            const int n = nt * 16 + lm;
            v8f acc = {};
            #pragma unroll
            for (int c = 0; c < 8; ++c) {         // K = 256 in chunks of 32
                // B layout: lane col = lm; v0..7 = K[16hs + 0..15] (pairs)
                Frag16 bfr;
                const unsigned short* wp = sWT + (size_t)n * NEMB + 32 * c + 16 * hs;
                bfr.q[0] = *(const uint4*)wp;
                bfr.q[1] = *(const uint4*)(wp + 8);
                acc = wmma_bf16(xa[c], bfr, acc);
            }
            if (w < 2) {
                // q/k row-major: C frag row M = r + 8hs -> global row R+8hs+r
                unsigned short* dst = (w == 0) ? qo : ko;
                #pragma unroll
                for (int r = 0; r < 8; ++r)
                    dst[(size_t)(R + 8 * hs + r) * HS + n] = f2bf(acc[r]);
            } else {
                // V transposed: 8 consecutive t for fixed h -> one b128 store
                uint4 pk;
                pk.x = pkbf(acc[0], acc[1]);
                pk.y = pkbf(acc[2], acc[3]);
                pk.z = pkbf(acc[4], acc[5]);
                pk.w = pkbf(acc[6], acc[7]);
                *(uint4*)(vto + ((size_t)bb * HS + n) * SEQ + t0 + 8 * hs) = pk;
            }
        }
    }
}

// ---------------------------------------------------------------------------
// Kernel 2: causal softmax attention with dropout.
// One workgroup per batch (256 threads = 8 waves); K and V^T in LDS.
// Each wave handles m-tiles {w, 15-w} -> uniform 17 s-tiles per wave.
// Computes S^T = K * Q^T so the C-fragment lane = column m; softmax reduction
// is in-lane + one shfl_xor(16); P^T C-layout repacks directly into the
// B-operand of O^T = V^T * P^T (no LDS round trip for P).
// ---------------------------------------------------------------------------
__global__ __launch_bounds__(256) void attn_kernel(
    const unsigned short* __restrict__ q,   // [B*T][64] bf16
    const unsigned short* __restrict__ k,   // [B*T][64] bf16
    const unsigned short* __restrict__ vt,  // [B][64][T] bf16
    const float* __restrict__ dm,           // dropout noise [B][T][T]
    float* __restrict__ out)                // [B][T][64] f32
{
    __shared__ unsigned short sK[SEQ * HS];    // 32 KB  [s][d]
    __shared__ unsigned short sVT[HS * SEQ];   // 32 KB  [d][s]

    const int tid  = threadIdx.x;
    const int lane = tid & 31;
    const int wave = tid >> 5;
    const int hs   = lane >> 4;
    const int lm   = lane & 15;
    const int b    = blockIdx.x;

    {   // cooperative staging of K and V^T (bf16, b128 copies)
        const uint4* gk = (const uint4*)(k  + (size_t)b * SEQ * HS);
        const uint4* gv = (const uint4*)(vt + (size_t)b * SEQ * HS);
        uint4* lk = (uint4*)sK;
        uint4* lv = (uint4*)sVT;
        #pragma unroll
        for (int i = 0; i < (SEQ * HS / 8) / 256; ++i) {
            lk[tid + 256 * i] = gk[tid + 256 * i];
            lv[tid + 256 * i] = gv[tid + 256 * i];
        }
    }
    __syncthreads();

    #pragma unroll
    for (int half = 0; half < 2; ++half) {
        const int mt = half ? (15 - wave) : wave;   // m-tile index 0..15
        const int m  = mt * 16 + lm;                // this lane's query row

        // Q as B-fragments for S^T = K * Q^T  (d 0..31 and 32..63)
        Frag16 qb0, qb1;
        const unsigned short* qrow = q + ((size_t)b * SEQ + m) * HS;
        qb0.q[0] = *(const uint4*)(qrow + 16 * hs);
        qb0.q[1] = *(const uint4*)(qrow + 16 * hs + 8);
        qb1.q[0] = *(const uint4*)(qrow + 32 + 16 * hs);
        qb1.q[1] = *(const uint4*)(qrow + 32 + 16 * hs + 8);

        v8f oc[4] = {};            // O^T accumulators, 4 d-tiles of 16
        float lsum = 0.0f;         // softmax denominator (this lane's s subset)
        const float* dmrow = dm + ((size_t)b * SEQ + m) * SEQ;

        const int nchunks = mt / 2 + 1;            // causal: skip tiles > mt
        for (int c = 0; c < nchunks; ++c) {
            // prefetch next chunk of the dropout-noise row (streaming read)
            if (c + 1 < nchunks)
                __builtin_prefetch(dmrow + (2 * c + 2) * 16 + 8 * hs, 0, 0);

            unsigned pk01[2][4];
            #pragma unroll
            for (int ht = 0; ht < 2; ++ht) {
                const int t = 2 * c + ht;          // s-tile index
                unsigned* pk = pk01[ht];
                if (t > mt) { pk[0] = pk[1] = pk[2] = pk[3] = 0u; continue; }

                // A = K tile from LDS (rows s, K-dim d)
                const unsigned short* kp = sK + (size_t)(t * 16 + lm) * HS;
                Frag16 a0, a1;
                a0.q[0] = *(const uint4*)(kp + 16 * hs);
                a0.q[1] = *(const uint4*)(kp + 16 * hs + 8);
                a1.q[0] = *(const uint4*)(kp + 32 + 16 * hs);
                a1.q[1] = *(const uint4*)(kp + 32 + 16 * hs + 8);

                v8f s = {};
                s = wmma_bf16(a0, qb0, s);
                s = wmma_bf16(a1, qb1, s);
                // S^T frag: lane col = m, rows s = t*16 + r + 8hs

                // dropout noise for this lane's 8 s values (contiguous)
                const float* mp = dmrow + t * 16 + 8 * hs;
                float4 mk0 = *(const float4*)mp;
                float4 mk1 = *(const float4*)(mp + 4);
                float mk[8] = {mk0.x, mk0.y, mk0.z, mk0.w,
                               mk1.x, mk1.y, mk1.z, mk1.w};

                const bool diag = (t == mt);
                float p[8];
                #pragma unroll
                for (int r = 0; r < 8; ++r) {
                    // exp(s/8) == exp2(s * 0.125*log2(e)) : one mul + v_exp_f32
                    float e = fast_exp2(s[r] * SCALE_LOG2E);
                    if (diag && (r + 8 * hs) > lm) e = 0.0f;   // causal mask
                    lsum += e;                            // denom pre-dropout
                    p[r] = (mk[r] >= DROP_P) ? e : 0.0f;  // dropout on P only
                }
                #pragma unroll
                for (int j = 0; j < 4; ++j) pk[j] = pkbf(p[2 * j], p[2 * j + 1]);
            }

            // Repack two S^T C-frags -> one B-fragment of P^T (K = 32 s)
            Frag16 pb;
            #pragma unroll
            for (int j = 0; j < 4; ++j) {
                unsigned send = hs ? pk01[0][j] : pk01[1][j];
                unsigned recv = (unsigned)__shfl_xor((int)send, 16, 32);
                pb.u[j]     = hs ? recv : pk01[0][j];
                pb.u[j + 4] = hs ? pk01[1][j] : recv;
            }

            // O^T += V^T(tile) * P^T(chunk)
            #pragma unroll
            for (int dt = 0; dt < 4; ++dt) {
                const unsigned short* vp =
                    sVT + (size_t)(dt * 16 + lm) * SEQ + c * 32;
                Frag16 va;
                va.q[0] = *(const uint4*)(vp + 16 * hs);
                va.q[1] = *(const uint4*)(vp + 16 * hs + 8);
                oc[dt] = wmma_bf16(va, pb, oc[dt]);
            }
        }

        // normalize: 1/l over both half-wave s subsets, fold in dropout 1/(1-p)
        float lfull = lsum + __shfl_xor(lsum, 16, 32);
        float sc = KEEP_INV / lfull;

        float* orow = out + ((size_t)b * SEQ + m) * HS;
        #pragma unroll
        for (int dt = 0; dt < 4; ++dt) {
            v8f o = oc[dt];
            float4 s0, s1;
            s0.x = o[0] * sc; s0.y = o[1] * sc; s0.z = o[2] * sc; s0.w = o[3] * sc;
            s1.x = o[4] * sc; s1.y = o[5] * sc; s1.z = o[6] * sc; s1.w = o[7] * sc;
            *(float4*)(orow + dt * 16 + 8 * hs)     = s0;
            *(float4*)(orow + dt * 16 + 8 * hs + 4) = s1;
        }
    }
}

// ---------------------------------------------------------------------------
extern "C" void kernel_launch(void* const* d_in, const int* in_sizes, int n_in,
                              void* d_out, int out_size, void* d_ws, size_t ws_size,
                              hipStream_t stream) {
    (void)in_sizes; (void)n_in; (void)out_size; (void)ws_size;
    const float* x  = (const float*)d_in[0];
    const float* Wq = (const float*)d_in[1];
    const float* Wk = (const float*)d_in[2];
    const float* Wv = (const float*)d_in[3];
    const float* dm = (const float*)d_in[4];
    float* out = (float*)d_out;

    const size_t nrows = (size_t)BATCH * SEQ;           // 131072
    unsigned short* qo = (unsigned short*)d_ws;         // 16 MB
    unsigned short* ko = qo + nrows * HS;               // 16 MB
    unsigned short* vt = ko + nrows * HS;               // 16 MB (V^T)
    unsigned short* wt = vt + nrows * HS;               // 96 KB (W^T bf16 x3)

    prep_kernel<<<dim3(3 * NEMB * HS / 256), dim3(256), 0, stream>>>(
        Wq, Wk, Wv, wt);
    qkv_kernel<<<dim3((unsigned)(nrows / 128)), dim3(256), 0, stream>>>(
        x, wt, qo, ko, vt);
    attn_kernel<<<dim3(BATCH), dim3(256), 0, stream>>>(qo, ko, vt, dm, out);
}
